// LMModel_RNN_67053029425812
// MI455X (gfx1250) — matrix operand order repaired and checked by reading, not compiled, not run on hardware
//
#include <hip/hip_runtime.h>

#define T_ 256
#define B_ 32
#define V_ 32000
#define H_ 256
#define L_ 2

typedef __attribute__((ext_vector_type(16))) __bf16        v16bf;
typedef __attribute__((ext_vector_type(8)))  float         v8f;
typedef __attribute__((ext_vector_type(4)))  unsigned int  v4u;

union FragBF { v4u q[2]; v16bf v; };

__device__ __forceinline__ unsigned short f2bf(float f) {
  unsigned int u = __float_as_uint(f);
  u += 0x7fffu + ((u >> 16) & 1u);          // round-to-nearest-even
  return (unsigned short)(u >> 16);
}

// A-matrix 16x32 bf16 fragment: lane half h gets 16 contiguous K starting at 16h
__device__ __forceinline__ v16bf load_fragA(const unsigned short* p, int half) {
  FragBF f;
  f.q[0] = *(const v4u*)(p + 16 * half);
  f.q[1] = *(const v4u*)(p + 16 * half + 8);
  return f.v;
}
// B-matrix 32x16 bf16 fragment: lane half h gets 16 contiguous K starting at 16h (col fixed)
__device__ __forceinline__ v16bf load_fragB(const unsigned short* p, int half) {
  FragBF f;
  const unsigned short* q = p + 16 * half;
  f.q[0] = *(const v4u*)(q);
  f.q[1] = *(const v4u*)(q + 8);
  return f.v;
}

// ---- prep: transpose + convert RNN weights to bf16: Wt[l][n][k] = W[l][k][n]
__global__ void k_prep_weights(const float* __restrict__ Wi, const float* __restrict__ Wh,
                               unsigned short* __restrict__ WiT, unsigned short* __restrict__ WhT) {
  int idx = blockIdx.x * blockDim.x + threadIdx.x;   // < L*H*H
  int l = idx / (H_ * H_);
  int r = idx - l * H_ * H_;
  int k = r / H_;
  int n = r - k * H_;
  int dst = l * H_ * H_ + n * H_ + k;
  WiT[dst] = f2bf(Wi[idx]);
  WhT[dst] = f2bf(Wh[idx]);
}

// ---- prep: decW (row-major [V][H]) -> bf16, layout already matches B-fragment needs
__global__ void k_prep_decw(const float* __restrict__ decW, unsigned short* __restrict__ decWbf) {
  int idx = blockIdx.x * blockDim.x + threadIdx.x;   // < V*H
  decWbf[idx] = f2bf(decW[idx]);
}

// ---- prep: embedding gather to bf16 [T*B][H]
__global__ void k_embed(const int* __restrict__ ids, const float* __restrict__ emb,
                        unsigned short* __restrict__ Ebf) {
  int idx = blockIdx.x * blockDim.x + threadIdx.x;   // < T*B*H
  int row = idx >> 8;
  int k = idx & 255;
  int id = ids[row];
  Ebf[idx] = f2bf(emb[(size_t)id * H_ + k]);
}

// ---- parallel precompute: X0 = E @ Wi[0] + bh[0]   (removes layer-0 input GEMM
//      from the sequential chain).  M=8192, N=256, K=256.
__global__ __launch_bounds__(256) void k_xgemm(
    const unsigned short* __restrict__ Ebf,    // [8192][256] bf16
    const unsigned short* __restrict__ WiT0,   // [256(n)][256(k)] bf16
    const float* __restrict__ bh,              // bh[0][*]
    float* __restrict__ X0)                    // [8192][256] f32
{
  const int lane = threadIdx.x & 31;
  const int wave = threadIdx.x >> 5;           // 0..7, 2 N-tiles each
  const int l15  = lane & 15;
  const int half = lane >> 4;
  const int i0 = blockIdx.x * 16;
  const int n0 = wave * 32;

  const unsigned short* Arow = Ebf + (size_t)(i0 + l15) * H_;
  v8f acc[2];
#pragma unroll
  for (int j = 0; j < 2; ++j)
    acc[j] = (v8f){0.f, 0.f, 0.f, 0.f, 0.f, 0.f, 0.f, 0.f};

#pragma unroll
  for (int k0 = 0; k0 < H_; k0 += 32) {
    v16bf a = load_fragA(Arow + k0, half);
#pragma unroll
    for (int j = 0; j < 2; ++j) {
      v16bf b = load_fragB(WiT0 + (size_t)(n0 + j * 16 + l15) * H_ + k0, half);
      acc[j] = __builtin_amdgcn_wmma_f32_16x16x32_bf16(false, a, false, b, (short)0, acc[j], false, false);
    }
  }
#pragma unroll
  for (int j = 0; j < 2; ++j) {
    int col = n0 + j * 16 + l15;
    float bias = bh[col];
#pragma unroll
    for (int v = 0; v < 8; ++v)
      X0[(size_t)(i0 + v + 8 * half) * H_ + col] = acc[j][v] + bias;
  }
}

// ---- sequential RNN: one workgroup, 32 waves, one 16x16 tile per wave.
// Wh0 + Wi1 live in LDS (256 KB, immune to LICM spilling); each wave keeps its
// 16-column slice of Wh1 as prebuilt WMMA B-fragments in 64 VGPRs.
__global__ __launch_bounds__(1024) void k_rnn(
    const float* __restrict__ X0,             // [T*B][H] f32 = E@Wi0 + bh0
    const unsigned short* __restrict__ WhT0g, // [256][256] bf16 (n-major)
    const unsigned short* __restrict__ WiT1g, // [256][256] bf16
    const unsigned short* __restrict__ WhT1g, // [256][256] bf16
    const float* __restrict__ bh,             // [L][H]
    unsigned short* __restrict__ outbf,       // [T*B][H] bf16 (decoder A)
    float* __restrict__ hiddenOut)            // [L][B][H] f32
{
  extern __shared__ __align__(16) unsigned short smem[];
  unsigned short* Wh0L = smem;                         // 65536 elems (128 KB)
  unsigned short* Wi1L = smem + 65536;                 // 65536 elems (128 KB)
  unsigned short* hbf  = smem + 131072;                // [2][32][256]  (32 KB)

  // cooperative LDS fill: 256 KB of weights (b128 copies) + zero hidden state
  {
    const v4u* s0 = (const v4u*)WhT0g;
    const v4u* s1 = (const v4u*)WiT1g;
    v4u* d0 = (v4u*)Wh0L;
    v4u* d1 = (v4u*)Wi1L;
    for (int i = threadIdx.x; i < 65536 / 8; i += 1024) { d0[i] = s0[i]; d1[i] = s1[i]; }
    for (int i = threadIdx.x; i < L_ * B_ * H_; i += 1024) hbf[i] = 0;
  }

  const int lane = threadIdx.x & 31;
  const int wave = threadIdx.x >> 5;     // 0..31
  const int mt   = wave & 1;
  const int nt   = wave >> 1;            // 0..15
  const int l15  = lane & 15;
  const int half = lane >> 4;
  const int arow = mt * 16 + l15;        // A-operand row (batch)
  const int ncol = nt * 16 + l15;        // output column (hidden unit)

  // deliberately hoist this wave's Wh1 column-slice into registers (B fragments)
  FragBF wh1[8];
  {
    const unsigned short* Brow = WhT1g + (size_t)ncol * H_;
#pragma unroll
    for (int ks = 0; ks < 8; ++ks) {
      const unsigned short* q = Brow + ks * 32 + 16 * half;
      wh1[ks].q[0] = *(const v4u*)(q);
      wh1[ks].q[1] = *(const v4u*)(q + 8);
    }
  }
  const float bias1 = bh[H_ + ncol];

  __syncthreads();

  for (int t = 0; t < T_; ++t) {
    // ---- layer 0: h0 = sigmoid(X0[t] + h0 @ Wh0)
    v8f acc;
#pragma unroll
    for (int v = 0; v < 8; ++v)
      acc[v] = X0[(size_t)(t * B_ + mt * 16 + v + 8 * half) * H_ + ncol];

#pragma unroll
    for (int k0 = 0; k0 < H_; k0 += 32) {
      v16bf a = load_fragA(hbf + arow * H_ + k0, half);           // h0 (LDS)
      v16bf b = load_fragB(Wh0L + ncol * H_ + k0, half);          // Wh0 (LDS)
      acc = __builtin_amdgcn_wmma_f32_16x16x32_bf16(false, a, false, b, (short)0, acc, false, false);
    }
    float s0[8];
#pragma unroll
    for (int v = 0; v < 8; ++v)
      s0[v] = 1.0f / (1.0f + __expf(-acc[v]));

    __syncthreads();   // all reads of hbf[0] done before overwrite
#pragma unroll
    for (int v = 0; v < 8; ++v) {
      int row = mt * 16 + v + 8 * half;
      hbf[row * H_ + ncol] = f2bf(s0[v]);
      if (t == T_ - 1) hiddenOut[(size_t)row * H_ + ncol] = s0[v];
    }
    __syncthreads();

    // ---- layer 1: h1 = sigmoid(h0 @ Wi1 + h1 @ Wh1 + bh1)
#pragma unroll
    for (int v = 0; v < 8; ++v) acc[v] = 0.f;
#pragma unroll
    for (int k0 = 0; k0 < H_; k0 += 32) {
      v16bf a = load_fragA(hbf + arow * H_ + k0, half);           // h0 (LDS)
      v16bf b = load_fragB(Wi1L + ncol * H_ + k0, half);          // Wi1 (LDS)
      acc = __builtin_amdgcn_wmma_f32_16x16x32_bf16(false, a, false, b, (short)0, acc, false, false);
    }
#pragma unroll
    for (int ks = 0; ks < 8; ++ks) {
      v16bf a = load_fragA(hbf + B_ * H_ + arow * H_ + ks * 32, half);  // h1 (LDS)
      acc = __builtin_amdgcn_wmma_f32_16x16x32_bf16(false, a, false, wh1[ks].v, (short)0, acc, false, false);
    }
    float s1[8];
#pragma unroll
    for (int v = 0; v < 8; ++v)
      s1[v] = 1.0f / (1.0f + __expf(-(acc[v] + bias1)));

    __syncthreads();   // all reads of hbf[1] done before overwrite
#pragma unroll
    for (int v = 0; v < 8; ++v) {
      int row = mt * 16 + v + 8 * half;
      unsigned short bv = f2bf(s1[v]);
      hbf[B_ * H_ + row * H_ + ncol] = bv;
      outbf[(size_t)t * B_ * H_ + row * H_ + ncol] = bv;
      if (t == T_ - 1) hiddenOut[(size_t)B_ * H_ + row * H_ + ncol] = s1[v];
    }
    __syncthreads();
  }
}

// ---- decoder GEMM: [8192 x 256] @ [256 x 32000], each wave does 16x80
__global__ __launch_bounds__(256) void k_decode(
    const unsigned short* __restrict__ outbf,    // [8192][256] bf16
    const unsigned short* __restrict__ decWbf,   // [32000][256] bf16
    const float* __restrict__ decB,              // [32000]
    float* __restrict__ decoded)                 // [8192][32000] f32
{
  const int lane = threadIdx.x & 31;
  const int wave = threadIdx.x >> 5;             // 0..7
  const int l15  = lane & 15;
  const int half = lane >> 4;
  const int i0 = blockIdx.y * 16;
  const int n0 = blockIdx.x * 640 + wave * 80;   // 5 N-tiles per wave

  const unsigned short* Arow = outbf + (size_t)(i0 + l15) * H_;

  v8f acc[5];
#pragma unroll
  for (int j = 0; j < 5; ++j)
    acc[j] = (v8f){0.f, 0.f, 0.f, 0.f, 0.f, 0.f, 0.f, 0.f};

#pragma unroll
  for (int k0 = 0; k0 < H_; k0 += 32) {
    v16bf a = load_fragA(Arow + k0, half);
#pragma unroll
    for (int j = 0; j < 5; ++j) {
      const unsigned short* Brow = decWbf + (size_t)(n0 + j * 16 + l15) * H_ + k0;
      v16bf b = load_fragB(Brow, half);
      acc[j] = __builtin_amdgcn_wmma_f32_16x16x32_bf16(false, a, false, b, (short)0, acc[j], false, false);
    }
  }

#pragma unroll
  for (int j = 0; j < 5; ++j) {
    int col = n0 + j * 16 + l15;
    float bias = decB[col];
#pragma unroll
    for (int v = 0; v < 8; ++v) {
      int row = i0 + v + 8 * half;
      __builtin_nontemporal_store(acc[j][v] + bias, decoded + (size_t)row * V_ + col);
    }
  }
}

extern "C" void kernel_launch(void* const* d_in, const int* in_sizes, int n_in,
                              void* d_out, int out_size, void* d_ws, size_t ws_size,
                              hipStream_t stream) {
  (void)in_sizes; (void)n_in; (void)out_size; (void)ws_size;
  const int*   ids  = (const int*)d_in[0];
  const float* emb  = (const float*)d_in[1];
  const float* Wi   = (const float*)d_in[2];
  const float* Wh   = (const float*)d_in[3];
  const float* bh   = (const float*)d_in[4];
  const float* decW = (const float*)d_in[5];
  const float* decB = (const float*)d_in[6];
  float* out = (float*)d_out;

  // workspace layout (bytes); total ~33.7 MB
  char* ws = (char*)d_ws;
  unsigned short* Ebf    = (unsigned short*)(ws + 0);          //  4,194,304
  unsigned short* outbf  = (unsigned short*)(ws + 4194304);    //  4,194,304
  unsigned short* WiT    = (unsigned short*)(ws + 8388608);    //    262,144
  unsigned short* WhT    = (unsigned short*)(ws + 8650752);    //    262,144
  unsigned short* decWbf = (unsigned short*)(ws + 8912896);    // 16,384,000
  float*          X0     = (float*)(ws + 25296896);            //  8,388,608

  k_prep_weights<<<dim3((L_ * H_ * H_) / 256), dim3(256), 0, stream>>>(Wi, Wh, WiT, WhT);
  k_prep_decw  <<<dim3((V_ * H_) / 256),       dim3(256), 0, stream>>>(decW, decWbf);
  k_embed      <<<dim3((T_ * B_ * H_) / 256),  dim3(256), 0, stream>>>(ids, emb, Ebf);

  // parallel non-recurrent GEMM: X0 = E @ Wi0 + bh0
  k_xgemm<<<dim3((T_ * B_) / 16), dim3(256), 0, stream>>>(Ebf, WiT, bh, X0);

  // sequential RNN with 288 KB dynamic LDS (Wh0 + Wi1 + hidden state)
  const int rnnLds = 2 * 65536 * 2 + L_ * B_ * H_ * 2;   // 294912 B
  (void)hipFuncSetAttribute((const void*)k_rnn,
                            hipFuncAttributeMaxDynamicSharedMemorySize, rnnLds);
  k_rnn<<<dim3(1), dim3(1024), rnnLds, stream>>>(
      X0, WhT, WiT + H_ * H_, WhT + H_ * H_, bh, outbf, out + (size_t)T_ * B_ * V_);

  // 50 * 640 = 32000 cols, 512 * 16 = 8192 rows — exact cover
  k_decode<<<dim3(50, 512), dim3(256), 0, stream>>>(outbf, decWbf, decB, out);
}
